// WindowAttention_19069654794851
// MI455X (gfx1250) — matrix operand order repaired and checked by reading, not compiled, not run on hardware
//
#include <hip/hip_runtime.h>
#include <hip/hip_bf16.h>

typedef __attribute__((ext_vector_type(16))) _Float16 v16h;
typedef __attribute__((ext_vector_type(8)))  float    v8f;

#define HEADS 12
#define HD    32
#define NQ    49
#define NP    64
#define DIMC  384
#define KT    12                     // 384 / 32
#define NWIN  64
#define QK_SCALE 0.17677669529663687f   // 32^-0.5

// ---------------- workspace layout (bytes) ----------------
#define WQKV_ELEMS  (12*3*2*12*512)          // frag-ordered f16 qkv weights
#define WPROJ_ELEMS (24*12*512)              // frag-ordered f16 proj weights
#define WQKV_OFF    0
#define WPROJ_OFF   (WQKV_OFF + WQKV_ELEMS*2)
#define BIASC_OFF   (WPROJ_OFF + WPROJ_ELEMS*2)
#define BIASC_N     (NWIN*HEADS*64*64)       // combined mask+rel-bias, padded
#define BQKV_OFF    (BIASC_OFF + BIASC_N*4)
#define BQKV_N      (3*384)

// ---------------- LDS layout (bytes) ----------------
#define LDS_XH      0                        // [12][64][32] f16 (X, later O)
#define LDS_WAVE    (12*64*32*2)             // 49152
#define WAVE_BYTES  14336
#define QH_OFF      0                        // [64][32] f16
#define KH_OFF      4096                     // [64][32] f16
#define VT_OFF      8192                     // [32][64] f16 (transposed V)
#define PH_OFF      12288                    // [2][16][32] f16
#define SMEM_TOTAL  (LDS_WAVE + 12*WAVE_BYTES)   // 221184 B = 216 KB

union F16Frag { v16h v; uint4 q[2]; };

// A fragment (16x32 f16) from LDS tile laid out [64 rows][32 k].
// Lane L: M = L&15 (+ 16*mt), K = 16*(L>>4) + e
__device__ __forceinline__ v16h ldsA(const _Float16* base, int mt, int lane) {
  int row = mt * 16 + (lane & 15);
  int hi  = lane >> 4;
  F16Frag f;
  f.q[0] = *(const uint4*)(base + row * 32 + 16 * hi);
  f.q[1] = *(const uint4*)(base + row * 32 + 16 * hi + 8);
  return f.v;
}

// B fragment (32x16 f16): lane L holds column n = L&15, K = 16*(L>>4) + e,
// i.e. 16 contiguous halves at p.
__device__ __forceinline__ v16h ldB16(const _Float16* p) {
  F16Frag f;
  f.q[0] = *(const uint4*)(p);
  f.q[1] = *(const uint4*)(p + 8);
  return f.v;
}

__device__ __forceinline__ v8f wmma16(v16h a, v16h b, v8f c) {
  return __builtin_amdgcn_wmma_f32_16x16x32_f16(false, a, false, b,
                                                (short)0, c, false, false);
}

// =================== prep kernels ===================

// qkv_w [384][1152] -> B-fragment-ordered f16, SCALE folded into q part.
// frag f = ((head*3+mat)*2+nt)*12+kt ; element (lane,e):
//   W[kt*32 + 16*(lane>>4) + e][mat*384 + head*32 + nt*16 + (lane&15)]
__global__ void prep_wqkv(const float* __restrict__ qkv_w,
                          _Float16* __restrict__ wq) {
  int tid = blockIdx.x * 256 + threadIdx.x;
  if (tid >= WQKV_ELEMS) return;
  int e    = tid & 15;
  int lane = (tid >> 4) & 31;
  int f    = tid >> 9;
  int kt   = f % 12;
  int nt   = (f / 12) & 1;
  int mat  = (f / 24) % 3;
  int h    = f / 72;
  int krow = kt * 32 + 16 * (lane >> 4) + e;
  int col  = mat * DIMC + h * HD + nt * 16 + (lane & 15);
  float v = qkv_w[krow * (3 * DIMC) + col];
  if (mat == 0) v *= QK_SCALE;
  wq[tid] = (_Float16)v;
}

__global__ void prep_wproj(const float* __restrict__ proj_w,
                           _Float16* __restrict__ wp) {
  int tid = blockIdx.x * 256 + threadIdx.x;
  if (tid >= WPROJ_ELEMS) return;
  int e    = tid & 15;
  int lane = (tid >> 4) & 31;
  int f    = tid >> 9;
  int kt   = f % 12;
  int ntg  = f / 12;                 // 0..23
  int krow = kt * 32 + 16 * (lane >> 4) + e;
  int col  = ntg * 16 + (lane & 15);
  wp[tid] = (_Float16)proj_w[krow * DIMC + col];
}

// biasC[win][h][64][64] = rel_bias[h][i][j] + mask[win][i][j], padded:
//   j >= 49 -> -1e30 (softmax kills padded keys), i >= 49 -> 0 (rows unused)
__global__ void prep_biasc(const float* __restrict__ tab,
                           const int* __restrict__ idx,
                           const float* __restrict__ mask,
                           float* __restrict__ biasC) {
  int tid = blockIdx.x * 256 + threadIdx.x;
  if (tid >= BIASC_N) return;
  int j   = tid & 63;
  int i   = (tid >> 6) & 63;
  int h   = (tid >> 12) % HEADS;
  int win = tid / (HEADS * 64 * 64);
  float v;
  if (j >= NQ)      v = -1.0e30f;
  else if (i >= NQ) v = 0.0f;
  else              v = tab[idx[i * NQ + j] * HEADS + h] +
                        mask[(win * NQ + i) * NQ + j];
  biasC[tid] = v;
}

__global__ void prep_bqkv(const float* __restrict__ qkv_b,
                          float* __restrict__ bq) {
  int tid = blockIdx.x * 256 + threadIdx.x;
  if (tid >= BQKV_N) return;
  float v = qkv_b[tid];
  if (tid < DIMC) v *= QK_SCALE;     // fold softmax scale into q bias
  bq[tid] = v;
}

// =================== fused window-attention kernel ===================
// 1 block per window (4096), 12 waves (= heads), wave32.
__global__ __launch_bounds__(384) void swin_attn_kernel(
    const float* __restrict__ x,
    const float* __restrict__ proj_b,
    const _Float16* __restrict__ Wqkv16,
    const _Float16* __restrict__ Wproj16,
    const float* __restrict__ biasC,
    const float* __restrict__ Bqkv,
    float* __restrict__ out) {
  extern __shared__ char smem[];
  _Float16* XhF = (_Float16*)(smem + LDS_XH);   // [12][64][32] f16

  const int tid  = threadIdx.x;
  const int lane = tid & 31;
  const int w    = tid >> 5;                    // wave id == head id
  const int hi   = lane >> 4;
  const int l15  = lane & 15;
  const int blk  = blockIdx.x;
  const int win  = blk & (NWIN - 1);            // window idx (nW = 64)

  _Float16* Qh = (_Float16*)(smem + LDS_WAVE + w * WAVE_BYTES + QH_OFF);
  _Float16* Kh = (_Float16*)(smem + LDS_WAVE + w * WAVE_BYTES + KH_OFF);
  _Float16* Vt = (_Float16*)(smem + LDS_WAVE + w * WAVE_BYTES + VT_OFF);
  _Float16* Ph = (_Float16*)(smem + LDS_WAVE + w * WAVE_BYTES + PH_OFF);

  // ---- stage 0: X -> LDS f16, layout [kt][64 rows][32 k], pad rows with 0
  {
    const float4* x4 = (const float4*)(x + (size_t)blk * NQ * DIMC);
    for (int i = tid; i < (NP * DIMC) / 4; i += 384) {
      int row = i / (DIMC / 4);
      int kq  = i % (DIMC / 4);
      int k   = kq * 4;
      float4 vv;
      if (row < NQ) vv = x4[row * (DIMC / 4) + kq];
      else { vv.x = 0.f; vv.y = 0.f; vv.z = 0.f; vv.w = 0.f; }
      union { _Float16 h[4]; uint2 u; } pk;
      pk.h[0] = (_Float16)vv.x; pk.h[1] = (_Float16)vv.y;
      pk.h[2] = (_Float16)vv.z; pk.h[3] = (_Float16)vv.w;
      *(uint2*)(XhF + (k >> 5) * (NP * 32) + row * 32 + (k & 31)) = pk.u;
    }
  }
  __syncthreads();

  // ---- stage 1: QKV GEMM for head w  (Q pre-scaled via weights/bias)
  // mat kept as a rolled loop; store target is data-driven (base + strides),
  // so the epilogue is branchless without tripling code size / VGPR pressure.
  #pragma unroll 1
  for (int mat = 0; mat < 3; ++mat) {
    _Float16* dbase = (mat == 0) ? Qh : (mat == 1) ? Kh : Vt;
    const int rs = (mat == 2) ? 1 : 32;      // row stride (Vt is transposed)
    const int cs = (mat == 2) ? NP : 1;      // col stride
    for (int mt = 0; mt < 4; ++mt) {
      for (int nt = 0; nt < 2; ++nt) {
        float bias = Bqkv[mat * DIMC + w * HD + nt * 16 + l15];
        v8f acc;
        #pragma unroll
        for (int r = 0; r < 8; ++r) acc[r] = bias;
        #pragma unroll
        for (int kt = 0; kt < KT; ++kt) {
          v16h a  = ldsA(XhF + kt * (NP * 32), mt, lane);
          int  f  = ((w * 3 + mat) * 2 + nt) * KT + kt;
          v16h bb = ldB16(Wqkv16 + f * 512 + lane * 16);
          acc = wmma16(a, bb, acc);
        }
        #pragma unroll
        for (int r = 0; r < 8; ++r) {
          int row = mt * 16 + r + 8 * hi;      // C layout: M = r + 8*(L>>4)
          int col = nt * 16 + l15;             // N = L&15
          dbase[row * rs + col * cs] = (_Float16)acc[r];
        }
      }
    }
  }
  __syncthreads();   // all waves done reading XhF; O can overwrite it below

  // ---- stage 2: attention for head w (head_dim 32 == one WMMA K-step)
  const float* biasH = biasC + ((size_t)(win * HEADS + w) << 12);  // [64][64]
  for (int mt = 0; mt < 4; ++mt) {
    v16h aq = ldsA(Qh, mt, lane);
    v8f s[4];
    #pragma unroll
    for (int nt = 0; nt < 4; ++nt) {
      v8f z;
      #pragma unroll
      for (int r = 0; r < 8; ++r) z[r] = 0.f;
      v16h bk = ldB16(Kh + (nt * 16 + l15) * 32 + 16 * hi);
      s[nt] = wmma16(aq, bk, z);
    }
    // bias + softmax (row's 64 cols live on the 16 lanes of a half-wave)
    #pragma unroll
    for (int r = 0; r < 8; ++r) {
      int i = mt * 16 + r + 8 * hi;
      float m = -3.0e38f;
      #pragma unroll
      for (int nt = 0; nt < 4; ++nt) {
        int j = nt * 16 + l15;
        s[nt][r] += biasH[i * 64 + j];
        m = fmaxf(m, s[nt][r]);
      }
      #pragma unroll
      for (int off = 1; off < 16; off <<= 1)
        m = fmaxf(m, __shfl_xor(m, off, 32));
      float sum = 0.f;
      #pragma unroll
      for (int nt = 0; nt < 4; ++nt) {
        float e = __expf(s[nt][r] - m);
        s[nt][r] = e;
        sum += e;
      }
      #pragma unroll
      for (int off = 1; off < 16; off <<= 1)
        sum += __shfl_xor(sum, off, 32);
      float inv = 1.f / sum;
      #pragma unroll
      for (int nt = 0; nt < 4; ++nt) {
        int j = nt * 16 + l15;
        Ph[(j >> 5) * 512 + (r + 8 * hi) * 32 + (j & 31)] =
            (_Float16)(s[nt][r] * inv);
      }
    }
    // P @ V  -> O rows of this mt, written into XhF slot [w]
    #pragma unroll
    for (int nt2 = 0; nt2 < 2; ++nt2) {
      v8f o;
      #pragma unroll
      for (int r = 0; r < 8; ++r) o[r] = 0.f;
      #pragma unroll
      for (int kt2 = 0; kt2 < 2; ++kt2) {
        v16h ap = ldsA(Ph + kt2 * 512, 0, lane);
        v16h bv = ldB16(Vt + (nt2 * 16 + l15) * NP + kt2 * 32 + 16 * hi);
        o = wmma16(ap, bv, o);
      }
      #pragma unroll
      for (int r = 0; r < 8; ++r) {
        int row = mt * 16 + r + 8 * hi;
        XhF[w * (NP * 32) + row * 32 + nt2 * 16 + l15] = (_Float16)o[r];
      }
    }
  }
  __syncthreads();

  // ---- stage 3: projection  out = O @ Wp + pb ; wave w owns n-tiles 2w,2w+1
  #pragma unroll 1
  for (int t = 0; t < 2; ++t) {
    int ntg = w * 2 + t;
    float pb = proj_b[ntg * 16 + l15];
    for (int mt = 0; mt < 4; ++mt) {
      v8f acc;
      #pragma unroll
      for (int r = 0; r < 8; ++r) acc[r] = pb;
      #pragma unroll
      for (int kt = 0; kt < KT; ++kt) {
        v16h a  = ldsA(XhF + kt * (NP * 32), mt, lane);
        v16h bb = ldB16(Wproj16 + (ntg * KT + kt) * 512 + lane * 16);
        acc = wmma16(a, bb, acc);
      }
      #pragma unroll
      for (int r = 0; r < 8; ++r) {
        int row = mt * 16 + r + 8 * hi;
        if (row < NQ)
          out[((size_t)blk * NQ + row) * DIMC + ntg * 16 + l15] = acc[r];
      }
    }
  }
}

extern "C" void kernel_launch(void* const* d_in, const int* in_sizes, int n_in,
                              void* d_out, int out_size, void* d_ws,
                              size_t ws_size, hipStream_t stream) {
  (void)in_sizes; (void)n_in; (void)out_size; (void)ws_size;
  const float* x       = (const float*)d_in[0];
  const float* mask    = (const float*)d_in[1];
  const float* qkv_w   = (const float*)d_in[2];
  const float* qkv_b   = (const float*)d_in[3];
  const float* proj_w  = (const float*)d_in[4];
  const float* proj_b  = (const float*)d_in[5];
  const float* rel_tab = (const float*)d_in[6];
  const int*   rel_idx = (const int*)d_in[7];
  float* out = (float*)d_out;

  char* ws = (char*)d_ws;
  _Float16* Wq16  = (_Float16*)(ws + WQKV_OFF);
  _Float16* Wp16  = (_Float16*)(ws + WPROJ_OFF);
  float*    biasC = (float*)(ws + BIASC_OFF);
  float*    Bq    = (float*)(ws + BQKV_OFF);

  prep_wqkv<<<(WQKV_ELEMS + 255) / 256, 256, 0, stream>>>(qkv_w, Wq16);
  prep_wproj<<<(WPROJ_ELEMS + 255) / 256, 256, 0, stream>>>(proj_w, Wp16);
  prep_biasc<<<(BIASC_N + 255) / 256, 256, 0, stream>>>(rel_tab, rel_idx,
                                                        mask, biasC);
  prep_bqkv<<<(BQKV_N + 255) / 256, 256, 0, stream>>>(qkv_b, Bq);

  (void)hipFuncSetAttribute(reinterpret_cast<const void*>(&swin_attn_kernel),
                            hipFuncAttributeMaxDynamicSharedMemorySize,
                            SMEM_TOTAL);
  swin_attn_kernel<<<4096, 384, SMEM_TOTAL, stream>>>(
      x, proj_b, Wq16, Wp16, biasC, Bq, out);
}
